// MultiHeadSelfAttention_83107617178033
// MI455X (gfx1250) — matrix operand order repaired and checked by reading, not compiled, not run on hardware
//
#include <hip/hip_runtime.h>
#include <hip/hip_bf16.h>

// ---------------------------------------------------------------------------
// MultiHeadSelfAttention on MI455X (gfx1250), compute-bound -> bf16 WMMA path.
//   B=4, L=1024, D=1024, H=16, DH=64.
//   v_wmma_f32_16x16x32_bf16 for QKV projection GEMMs and attention.
//   - S computed transposed (S^T = K * Q^T): softmax rows live in-lane and P
//     drops directly into the A-matrix VGPR layout of the PV WMMA.
//   - Ping-pong register buffers (unroll-by-2) in both hot loops: no buffer
//     rotation copies, loads overlap WMMA via partial s_wait_loadcnt.
// ---------------------------------------------------------------------------

typedef __attribute__((ext_vector_type(16))) __bf16 v16bf;
typedef __attribute__((ext_vector_type(8)))  __bf16 v8bf;
typedef __attribute__((ext_vector_type(4)))  __bf16 v4bf;
typedef __attribute__((ext_vector_type(8)))  float  v8f;

#define BB 4
#define LL 1024
#define DD 1024
#define HH 16
#define DH 64
#define NTOK (BB*LL)          // 4096
#define NQKV (3*DD)           // 3072
#define LOG2E 1.4426950408889634f

static __device__ __forceinline__ v8f wmma_bf16(v16bf a, v16bf b, v8f c) {
  // D = A(16x32 bf16) * B(32x16 bf16) + C(16x16 f32)
  return __builtin_amdgcn_wmma_f32_16x16x32_bf16(false, a, false, b, (short)0, c, false, false);
}

static __device__ __forceinline__ v16bf cat8(v8bf lo, v8bf hi) {
  return __builtin_shufflevector(lo, hi, 0,1,2,3,4,5,6,7,8,9,10,11,12,13,14,15);
}

// A-matrix fragment (ISA 16-bit A layout): row = ln, two 16B chunks 8*hl apart.
static __device__ __forceinline__ v16bf loadA_frag(const __bf16* row_base) {
  return cat8(*(const v8bf*)row_base, *(const v8bf*)(row_base + 16));
}

// -------------------------- f32 -> bf16 conversion -------------------------
__global__ __launch_bounds__(256) void cvt_kernel(
    const float* __restrict__ X, const float* __restrict__ Wq,
    const float* __restrict__ Wk, const float* __restrict__ Wv,
    __bf16* __restrict__ Xb, __bf16* __restrict__ Wb) {
  const int i = (blockIdx.x * blockDim.x + threadIdx.x) * 4;
  if (i < NTOK * DD) {
    const float4 x = *(const float4*)(X + i);
    v4bf p; p[0] = (__bf16)x.x; p[1] = (__bf16)x.y;
            p[2] = (__bf16)x.z; p[3] = (__bf16)x.w;
    *(v4bf*)(Xb + i) = p;
  }
  if (i < 3 * DD * DD) {
    const float* w = (i < DD*DD) ? Wq : (i < 2*DD*DD ? Wk : Wv);
    const float4 x = *(const float4*)(w + (i & (DD*DD - 1)));
    v4bf p; p[0] = (__bf16)x.x; p[1] = (__bf16)x.y;
            p[2] = (__bf16)x.z; p[3] = (__bf16)x.w;
    *(v4bf*)(Wb + i) = p;
  }
}

// ------------------------------ QKV projection -----------------------------
// Y = X @ Wc^T + bias,  Wc = [Wq;Wk;Wv] (3072 x 1024), X (4096 x 1024).
// One wave computes a 32x64 output tile: two A fragments share 4 B fragments
// (12 b128 loads per 8 WMMA). Ping-pong buffers over the K loop (no copies).
// Outputs: Q,K as bf16 (b,h,l,dh); V as bf16 transposed (b,h,dh,l).
__global__ __launch_bounds__(256) void qkv_gemm(
    const __bf16* __restrict__ X, const __bf16* __restrict__ W,
    const float* __restrict__ bq, const float* __restrict__ bk,
    const float* __restrict__ bv,
    __bf16* __restrict__ Qw, __bf16* __restrict__ Kw, __bf16* __restrict__ Vt) {
  const int wave = (blockIdx.x * blockDim.x + threadIdx.x) >> 5;
  const int lane = threadIdx.x & 31;
  const int hl = lane >> 4, ln = lane & 15;
  const int mt = wave / (NQKV / 64);          // 128 token tiles of 32
  const int ng = wave % (NQKV / 64);          // 48 groups of 64 features
  const int mbase = mt * 32, nbase = ng * 64;

  const __bf16* arow0 = X + (size_t)(mbase + ln) * DD + 8 * hl;
  const __bf16* arow1 = arow0 + (size_t)16 * DD;
  const __bf16* brow  = W + (size_t)(nbase + ln) * DD + 8 * hl;

  v8f acc[2][4] = {};

  auto loadChunk = [&](int c, v16bf& a0, v16bf& a1, v16bf (&Bv)[4]) {
    a0 = loadA_frag(arow0 + c * 32);
    a1 = loadA_frag(arow1 + c * 32);
#pragma unroll
    for (int t = 0; t < 4; ++t)
      Bv[t] = loadA_frag(brow + (size_t)t * 16 * DD + c * 32);
  };
  auto compute = [&](const v16bf& a0, const v16bf& a1, const v16bf (&Bv)[4]) {
#pragma unroll
    for (int t = 0; t < 4; ++t) acc[0][t] = wmma_bf16(a0, Bv[t], acc[0][t]);
#pragma unroll
    for (int t = 0; t < 4; ++t) acc[1][t] = wmma_bf16(a1, Bv[t], acc[1][t]);
  };

  v16bf Aa0, Aa1, Ba[4], Ab0, Ab1, Bb[4];
  loadChunk(0, Aa0, Aa1, Ba);
  for (int c = 0; c < DD / 32; c += 2) {      // 32 chunks (even count)
    loadChunk(c + 1, Ab0, Ab1, Bb);           // prefetch odd chunk
    compute(Aa0, Aa1, Ba);                    // compute even chunk
    if (c + 2 < DD / 32) loadChunk(c + 2, Aa0, Aa1, Ba);
    compute(Ab0, Ab1, Bb);                    // compute odd chunk
  }

  const int sec = nbase >> 10;                 // 0=Q 1=K 2=V (wave-uniform)
  const float* bias_ptr = (sec == 0) ? bq : (sec == 1) ? bk : bv;
  const int b = mbase >> 10;
#pragma unroll
  for (int s = 0; s < 2; ++s) {
    const int l0 = (mbase & (LL - 1)) + s * 16;
#pragma unroll
    for (int t = 0; t < 4; ++t) {
      const int f = (nbase + t * 16 + ln) & (DD - 1);   // feature within D
      const int h = f >> 6, dh = f & 63;
      const float bias = bias_ptr[f];
      if (sec < 2) {
        __bf16* dst = ((sec == 0) ? Qw : Kw) +
                      (((size_t)(b * HH + h) * LL + l0) * DH + dh);
#pragma unroll
        for (int r = 0; r < 8; ++r)
          dst[(size_t)(r + 8 * hl) * DH] = (__bf16)(acc[s][t][r] + bias);
      } else {
        // V transposed: lane holds 8 consecutive tokens -> one b128 store.
        v8bf pk;
#pragma unroll
        for (int r = 0; r < 8; ++r) pk[r] = (__bf16)(acc[s][t][r] + bias);
        *(v8bf*)(Vt + ((size_t)(b * HH + h) * DH + dh) * LL + l0 + 8 * hl) = pk;
      }
    }
  }
}

// ------------------------------- attention ---------------------------------
// One wave owns a 16-query tile of one (b,h). Flash-style online softmax over
// causal kv chunks of 32. Per chunk: 4 WMMA for S^T, 4 WMMA for PV.
// K fragments ping-pong across chunks; next-chunk K loads issue before the
// softmax VALU + PV WMMA block, which hides their latency.
__global__ __launch_bounds__(256) void attn_kernel(
    const __bf16* __restrict__ Qw, const __bf16* __restrict__ Kw,
    const __bf16* __restrict__ Vt, const float* __restrict__ pm,
    float* __restrict__ out) {
  const int wave = (blockIdx.x * blockDim.x + threadIdx.x) >> 5;
  const int lane = threadIdx.x & 31;
  const int hl = lane >> 4, ln = lane & 15;
  const int qt = wave & 63;                    // 64 query tiles per (b,h)
  const int bh = wave >> 6;                    // (b*H + h)
  const int b = bh >> 4;
  const int qbase = qt * 16;
  const int qrow = qbase + ln;                 // this lane's softmax row

  const __bf16* Qb = Qw + ((size_t)bh * LL + qbase + ln) * DH;
  const __bf16* Kr = Kw + (size_t)bh * LL * DH + (size_t)ln * DH + 8 * hl;
  const __bf16* Vr = Vt + (size_t)bh * DH * LL + (size_t)ln * LL + 8 * hl;
  const float* pmb = pm + b * LL;

  // Q resident: B-matrix layout of the S^T gemm (column ln = query row).
  v16bf qreg[2];
#pragma unroll
  for (int c = 0; c < 2; ++c)
    qreg[c] = loadA_frag(Qb + c * 32 + 8 * hl);

  v8f o[4] = {};
  float mprev = -3.0e38f, lsum = 0.0f;
  const float scale = 0.125f;                  // 1/sqrt(DH)
  const int nchunk = (qbase + 16 + 31) >> 5;   // causal kv coverage

  // K fragments: [0]=tile0 dh0..31, [1]=tile0 dh32..63, [2]/[3]=tile1.
  auto loadK = [&](int kb, v16bf (&K)[4]) {
    const __bf16* kp = Kr + (size_t)kb * DH;
    K[0] = loadA_frag(kp);
    K[1] = loadA_frag(kp + 32);
    K[2] = loadA_frag(kp + (size_t)16 * DH);
    K[3] = loadA_frag(kp + (size_t)16 * DH + 32);
  };

  auto chunk_body = [&](int ch, const v16bf (&Kc)[4], v16bf (&Kn)[4],
                        bool prefetch) {
    const int kb = ch * 32;
    v8f s0 = {}, s1 = {};
    s0 = wmma_bf16(Kc[0], qreg[0], s0);
    s0 = wmma_bf16(Kc[1], qreg[1], s0);
    s1 = wmma_bf16(Kc[2], qreg[0], s1);
    s1 = wmma_bf16(Kc[3], qreg[1], s1);

    // V fragments for this chunk, then next-chunk K: both issue before the
    // softmax VALU block so exp/mask math hides them.
    v16bf Bv[4];
#pragma unroll
    for (int t = 0; t < 4; ++t)
      Bv[t] = loadA_frag(Vr + (size_t)t * 16 * LL + kb);
    if (prefetch) loadK(kb + 32, Kn);

    // scale + padding mask + causal mask; local row max (row lives in-lane).
    float mloc = -3.0e38f;
#pragma unroll
    for (int r = 0; r < 8; ++r) {
      const int kv0 = kb + r + 8 * hl, kv1 = kv0 + 16;
      float v0 = fmaf(s0[r], scale, pmb[kv0]) + (kv0 > qrow ? -1.0e9f : 0.0f);
      float v1 = fmaf(s1[r], scale, pmb[kv1]) + (kv1 > qrow ? -1.0e9f : 0.0f);
      s0[r] = v0; s1[r] = v1;
      mloc = fmaxf(mloc, fmaxf(v0, v1));
    }
    mloc = fmaxf(mloc, __shfl_xor(mloc, 16, 32));
    const float mnew = fmaxf(mprev, mloc);
    const float alpha = __builtin_exp2f((mprev - mnew) * LOG2E);
    // exponentials; P lands directly in A-matrix bf16 layout (no transpose!)
    v16bf P;
    float rs = 0.0f;
#pragma unroll
    for (int r = 0; r < 8; ++r) {
      float e0 = __builtin_exp2f((s0[r] - mnew) * LOG2E);
      float e1 = __builtin_exp2f((s1[r] - mnew) * LOG2E);
      rs += e0 + e1;
      P[r] = (__bf16)e0; P[r + 8] = (__bf16)e1;
    }
    rs += __shfl_xor(rs, 16, 32);
    lsum = lsum * alpha + rs;
    mprev = mnew;
    // rescale O accumulators: row q = r + 8*hl -> broadcast alpha per VGPR.
#pragma unroll
    for (int r = 0; r < 8; ++r) {
      const float ar = __shfl(alpha, r + 8 * hl, 32);
      o[0][r] *= ar; o[1][r] *= ar; o[2][r] *= ar; o[3][r] *= ar;
    }
    // O += P @ V  (V^T rows are contiguous kv-major -> clean B fragments)
#pragma unroll
    for (int t = 0; t < 4; ++t) o[t] = wmma_bf16(P, Bv[t], o[t]);
  };

  v16bf KA[4], KB[4];
  loadK(0, KA);
  for (int ch = 0; ch < nchunk; ch += 2) {
    chunk_body(ch, KA, KB, ch + 1 < nchunk);
    if (ch + 1 < nchunk) chunk_body(ch + 1, KB, KA, ch + 2 < nchunk);
  }

  // normalize and store (out is (B, L, H*DH) f32)
  const float rinv = 1.0f / lsum;
  float* ob = out + ((size_t)b * LL + qbase) * DD + (bh & 15) * DH;
#pragma unroll
  for (int r = 0; r < 8; ++r) {
    const float sc = __shfl(rinv, r + 8 * hl, 32);
    const int row = r + 8 * hl;
#pragma unroll
    for (int t = 0; t < 4; ++t)
      ob[(size_t)row * DD + t * 16 + ln] = o[t][r] * sc;
  }
}

// ------------------------------- launcher ----------------------------------
extern "C" void kernel_launch(void* const* d_in, const int* in_sizes, int n_in,
                              void* d_out, int out_size, void* d_ws, size_t ws_size,
                              hipStream_t stream) {
  const float* seq = (const float*)d_in[0];
  const float* pm  = (const float*)d_in[1];
  // d_in[2] dependency_mask: causal, applied analytically in attn_kernel.
  const float* Wq  = (const float*)d_in[3];
  const float* bq  = (const float*)d_in[4];
  const float* Wk  = (const float*)d_in[5];
  const float* bk  = (const float*)d_in[6];
  const float* Wv  = (const float*)d_in[7];
  const float* bv  = (const float*)d_in[8];
  float* out = (float*)d_out;

  // workspace layout (bf16): X | [Wq;Wk;Wv] | Q | K | V^T   (~38 MB)
  __bf16* Xb = (__bf16*)d_ws;
  __bf16* Wb = Xb + (size_t)NTOK * DD;
  __bf16* Qb = Wb + (size_t)NQKV * DD;
  __bf16* Kb = Qb + (size_t)NTOK * DD;
  __bf16* Vt = Kb + (size_t)NTOK * DD;

  cvt_kernel<<<(NTOK * DD / 4 + 255) / 256, 256, 0, stream>>>(seq, Wq, Wk, Wv, Xb, Wb);

  // 128 M-tiles * 48 N-groups = 6144 waves, 8 waves/block
  qkv_gemm<<<6144 / 8, 256, 0, stream>>>(Xb, Wb, bq, bk, bv, Qb, Kb, Vt);

  // 4 * 16 * 64 = 4096 waves, 8 waves/block
  attn_kernel<<<4096 / 8, 256, 0, stream>>>(Qb, Kb, Vt, pm, out);
}
